// MultiHeadedAttention_76038101009012
// MI455X (gfx1250) — compile-verified, matrix operand order checked by reading, not run on hardware
//
#include <hip/hip_runtime.h>
#include <hip/hip_bf16.h>

typedef __attribute__((ext_vector_type(16))) _Float16 v16h;
typedef __attribute__((ext_vector_type(8)))  _Float16 v8h;
typedef __attribute__((ext_vector_type(8)))  float    v8f;

#define WMMA_F16(a, b, c) \
    __builtin_amdgcn_wmma_f32_16x16x32_f16(false, (a), false, (b), (short)0, (c), false, false)

static __device__ __forceinline__ v16h mk16(v8h lo, v8h hi) {
    return __builtin_shufflevector(lo, hi, 0,1,2,3,4,5,6,7,8,9,10,11,12,13,14,15);
}
// Fragment loader for the 16-bit WMMA A/B layout: lane holds row/col (lane&15),
// elements 0..7 = K kh..kh+7, elements 8..15 = K 16+kh..16+kh+7, kh=(lane>>4)*8.
static __device__ __forceinline__ v16h load_frag(const _Float16* __restrict__ p, int kh) {
    v8h lo = *(const v8h*)(p + kh);
    v8h hi = *(const v8h*)(p + 16 + kh);
    return mk16(lo, hi);
}

// ---------------------------------------------------------------- f32 -> f16
__global__ void cvt_f32_f16(const float* __restrict__ src, _Float16* __restrict__ dst, int n) {
    for (int i = blockIdx.x * blockDim.x + threadIdx.x; i < n; i += gridDim.x * blockDim.x)
        dst[i] = (_Float16)src[i];
}

// ------------------------------------------------- Y = X @ W^T + b, head-major out
// X: [8192 x 1024] f16 row-major (m = b*2048+s), W: [1024 x 1024] f16 row-major.
// out f16 at [((b*16+h)*2048 + s)*64 + d], n = h*64+d.
// Software-pipelined: fragments for k+32 are fetched while WMMAs consume k.
__global__ void __launch_bounds__(256)
proj_gemm(const _Float16* __restrict__ X, const _Float16* __restrict__ W,
          const float* __restrict__ bias, _Float16* __restrict__ out) {
    const int wave = threadIdx.x >> 5;
    const int lane = threadIdx.x & 31;
    const int strip = blockIdx.x * 8 + wave;      // 8192 strips: 512 M-tiles x 16 N-strips
    const int m0 = (strip >> 4) * 16;
    const int n0 = (strip & 15) * 64;
    const int r  = lane & 15;
    const int kh = (lane >> 4) * 8;

    const _Float16* arow = X + (size_t)(m0 + r) * 1024;
    const _Float16* w0 = W + (size_t)(n0 +  0 + r) * 1024;
    const _Float16* w1 = W + (size_t)(n0 + 16 + r) * 1024;
    const _Float16* w2 = W + (size_t)(n0 + 32 + r) * 1024;
    const _Float16* w3 = W + (size_t)(n0 + 48 + r) * 1024;

    v8f acc[4] = {v8f{}, v8f{}, v8f{}, v8f{}};
    // Prologue: fragments for k = 0.
    v16h a  = load_frag(arow, kh);
    v16h f0 = load_frag(w0, kh);
    v16h f1 = load_frag(w1, kh);
    v16h f2 = load_frag(w2, kh);
    v16h f3 = load_frag(w3, kh);
    for (int k0 = 32; k0 < 1024; k0 += 32) {
        // Prefetch next k-step's fragments before consuming current ones.
        v16h an  = load_frag(arow + k0, kh);
        v16h fn0 = load_frag(w0 + k0, kh);
        v16h fn1 = load_frag(w1 + k0, kh);
        v16h fn2 = load_frag(w2 + k0, kh);
        v16h fn3 = load_frag(w3 + k0, kh);
        acc[0] = WMMA_F16(a, f0, acc[0]);
        acc[1] = WMMA_F16(a, f1, acc[1]);
        acc[2] = WMMA_F16(a, f2, acc[2]);
        acc[3] = WMMA_F16(a, f3, acc[3]);
        a = an; f0 = fn0; f1 = fn1; f2 = fn2; f3 = fn3;
    }
    acc[0] = WMMA_F16(a, f0, acc[0]);
    acc[1] = WMMA_F16(a, f1, acc[1]);
    acc[2] = WMMA_F16(a, f2, acc[2]);
    acc[3] = WMMA_F16(a, f3, acc[3]);

    const int rbase = (lane < 16) ? 0 : 8;
#pragma unroll
    for (int t = 0; t < 4; ++t) {
        const int n = n0 + 16 * t + r;
        const float bv = bias[n];
        const int h = n >> 6, d = n & 63;
#pragma unroll
        for (int v = 0; v < 8; ++v) {
            const int m = m0 + rbase + v;
            const int b = m >> 11, s = m & 2047;
            out[(((size_t)(b * 16 + h)) * 2048 + s) * 64 + d] = (_Float16)(acc[t][v] + bv);
        }
    }
}

// ------------------------------------------------- flash attention, head-major q/k/v
// Block: 8 waves x 16 queries = 128 queries of one (b,h). Key tiles of 32.
__global__ void __launch_bounds__(256)
flash_attn(const _Float16* __restrict__ Q, const _Float16* __restrict__ K,
           const _Float16* __restrict__ V, _Float16* __restrict__ O) {
    __shared__ _Float16 vt[64 * 40];              // V^T tile: [d=64][key=32] stride 40
    const int wave = threadIdx.x >> 5;
    const int lane = threadIdx.x & 31;
    const int bh = blockIdx.x >> 4;               // b*16 + h
    const int qc = blockIdx.x & 15;
    const int c  = lane & 15;
    const int kh = (lane >> 4) * 8;
    const int q0 = qc * 128 + wave * 16;
    const size_t base = (size_t)bh * 2048 * 64;

    // Q as WMMA-B (K=d): lane holds query column q0+c, values contiguous over d.
    const _Float16* qrow = Q + base + (size_t)(q0 + c) * 64;
    const v16h qf0 = mk16(*(const v8h*)(qrow +      kh), *(const v8h*)(qrow + 16 + kh));
    const v16h qf1 = mk16(*(const v8h*)(qrow + 32 + kh), *(const v8h*)(qrow + 48 + kh));

    const float cs = 0.125f * 1.44269504088896340736f;  // 1/sqrt(64) * log2(e)
    float m2 = -3.0e38f, l = 0.0f;
    v8f oa[4] = {v8f{}, v8f{}, v8f{}, v8f{}};

    const int vkey = threadIdx.x >> 3;            // 0..31
    const int vd   = (threadIdx.x & 7) * 8;       // 0..56

    for (int kb = 0; kb < 2048; kb += 32) {
        // Stage V^T into LDS (transpose during store).
        v8h vrow = *(const v8h*)(V + base + (size_t)(kb + vkey) * 64 + vd);
#pragma unroll
        for (int j = 0; j < 8; ++j) vt[(vd + j) * 40 + vkey] = vrow[j];
        __syncthreads();

        // S^T = K_tile @ Q^T : two 16-key halves, K-dim = d (two steps of 32).
        const _Float16* krow0 = K + base + (size_t)(kb + c) * 64;
        const _Float16* krow1 = krow0 + 16 * 64;
        v8f st0{}, st1{};
        st0 = WMMA_F16(mk16(*(const v8h*)(krow0 +      kh), *(const v8h*)(krow0 + 16 + kh)), qf0, st0);
        st0 = WMMA_F16(mk16(*(const v8h*)(krow0 + 32 + kh), *(const v8h*)(krow0 + 48 + kh)), qf1, st0);
        st1 = WMMA_F16(mk16(*(const v8h*)(krow1 +      kh), *(const v8h*)(krow1 + 16 + kh)), qf0, st1);
        st1 = WMMA_F16(mk16(*(const v8h*)(krow1 + 32 + kh), *(const v8h*)(krow1 + 48 + kh)), qf1, st1);

        // Online softmax: lane holds 16 of query (lane&15)'s 32 scores; pair lane^16 has the rest.
        float mx = st0[0];
#pragma unroll
        for (int v = 1; v < 8; ++v) mx = fmaxf(mx, st0[v]);
#pragma unroll
        for (int v = 0; v < 8; ++v) mx = fmaxf(mx, st1[v]);
        mx = fmaxf(mx, __shfl_xor(mx, 16, 32));
        const float m2n   = fmaxf(m2, mx * cs);
        const float alpha = exp2f(m2 - m2n);
        float p0[8], p1[8], ls = 0.0f;
#pragma unroll
        for (int v = 0; v < 8; ++v) { p0[v] = exp2f(st0[v] * cs - m2n); ls += p0[v]; }
#pragma unroll
        for (int v = 0; v < 8; ++v) { p1[v] = exp2f(st1[v] * cs - m2n); ls += p1[v]; }
        ls += __shfl_xor(ls, 16, 32);
        l  = l * alpha + ls;
        m2 = m2n;
#pragma unroll
        for (int t = 0; t < 4; ++t) oa[t] = oa[t] * alpha;

        // P^T fragment (K=32 keys x N=16 queries): matches per-lane p layout exactly.
        v16h pf;
#pragma unroll
        for (int i = 0; i < 8; ++i) { pf[i] = (_Float16)p0[i]; pf[8 + i] = (_Float16)p1[i]; }

        // O^T += V^T @ P^T : 4 d-tiles of 16, K=32 keys.
#pragma unroll
        for (int t = 0; t < 4; ++t) {
            const int d = 16 * t + c;
            v16h vf = mk16(*(const v8h*)&vt[d * 40 + kh], *(const v8h*)&vt[d * 40 + 16 + kh]);
            oa[t] = WMMA_F16(vf, pf, oa[t]);
        }
        __syncthreads();
    }

    const float rl = 1.0f / l;
    const int rbase = (lane < 16) ? 0 : 8;
    const int b = bh >> 4, h = bh & 15;
#pragma unroll
    for (int t = 0; t < 4; ++t) {
        v8h ov;
#pragma unroll
        for (int v = 0; v < 8; ++v) ov[v] = (_Float16)(oa[t][v] * rl);
        const int d0 = 16 * t + rbase;
        *(v8h*)(O + (((size_t)b * 2048) + (q0 + c)) * 1024 + h * 64 + d0) = ov;
    }
}

// ------------------------------------------------- Y = O @ Wo^T + bo, f32 out
__global__ void __launch_bounds__(256)
out_gemm(const _Float16* __restrict__ X, const _Float16* __restrict__ W,
         const float* __restrict__ bias, float* __restrict__ Y) {
    const int wave = threadIdx.x >> 5;
    const int lane = threadIdx.x & 31;
    const int strip = blockIdx.x * 8 + wave;
    const int m0 = (strip >> 4) * 16;
    const int n0 = (strip & 15) * 64;
    const int r  = lane & 15;
    const int kh = (lane >> 4) * 8;

    const _Float16* arow = X + (size_t)(m0 + r) * 1024;
    const _Float16* w0 = W + (size_t)(n0 +  0 + r) * 1024;
    const _Float16* w1 = W + (size_t)(n0 + 16 + r) * 1024;
    const _Float16* w2 = W + (size_t)(n0 + 32 + r) * 1024;
    const _Float16* w3 = W + (size_t)(n0 + 48 + r) * 1024;

    v8f acc[4] = {v8f{}, v8f{}, v8f{}, v8f{}};
    v16h a  = load_frag(arow, kh);
    v16h f0 = load_frag(w0, kh);
    v16h f1 = load_frag(w1, kh);
    v16h f2 = load_frag(w2, kh);
    v16h f3 = load_frag(w3, kh);
    for (int k0 = 32; k0 < 1024; k0 += 32) {
        v16h an  = load_frag(arow + k0, kh);
        v16h fn0 = load_frag(w0 + k0, kh);
        v16h fn1 = load_frag(w1 + k0, kh);
        v16h fn2 = load_frag(w2 + k0, kh);
        v16h fn3 = load_frag(w3 + k0, kh);
        acc[0] = WMMA_F16(a, f0, acc[0]);
        acc[1] = WMMA_F16(a, f1, acc[1]);
        acc[2] = WMMA_F16(a, f2, acc[2]);
        acc[3] = WMMA_F16(a, f3, acc[3]);
        a = an; f0 = fn0; f1 = fn1; f2 = fn2; f3 = fn3;
    }
    acc[0] = WMMA_F16(a, f0, acc[0]);
    acc[1] = WMMA_F16(a, f1, acc[1]);
    acc[2] = WMMA_F16(a, f2, acc[2]);
    acc[3] = WMMA_F16(a, f3, acc[3]);

    const int rbase = (lane < 16) ? 0 : 8;
#pragma unroll
    for (int t = 0; t < 4; ++t) {
        const int n = n0 + 16 * t + r;
        const float bv = bias[n];
#pragma unroll
        for (int v = 0; v < 8; ++v) {
            const int m = m0 + rbase + v;
            Y[(size_t)m * 1024 + n] = acc[t][v] + bv;
        }
    }
}

extern "C" void kernel_launch(void* const* d_in, const int* in_sizes, int n_in,
                              void* d_out, int out_size, void* d_ws, size_t ws_size,
                              hipStream_t stream) {
    (void)in_sizes; (void)n_in; (void)out_size;
    const float* x  = (const float*)d_in[0];
    const float* Wq = (const float*)d_in[1]; const float* bq = (const float*)d_in[2];
    const float* Wk = (const float*)d_in[3]; const float* bk = (const float*)d_in[4];
    const float* Wv = (const float*)d_in[5]; const float* bv = (const float*)d_in[6];
    const float* Wo = (const float*)d_in[7]; const float* bo = (const float*)d_in[8];

    char* ws = (char*)d_ws;
    const size_t NX = 8388608;   // B*S*H
    const size_t NW = 1048576;   // H*H
    _Float16* x16  = (_Float16*)(ws);
    _Float16* wq16 = (_Float16*)(ws + 16777216);
    _Float16* wk16 = (_Float16*)(ws + 16777216 + 2097152);
    _Float16* wv16 = (_Float16*)(ws + 16777216 + 2 * 2097152);
    _Float16* wo16 = (_Float16*)(ws + 16777216 + 3 * 2097152);
    _Float16* q16  = (_Float16*)(ws + 25165824);
    _Float16* k16  = (_Float16*)(ws + 25165824 + 16777216);
    _Float16* v16  = (_Float16*)(ws + 25165824 + 2 * 16777216);
    _Float16* o16  = (_Float16*)(ws + 25165824 + 3 * 16777216);
    if (ws_size < (size_t)92274688) return;  // need ~92.3 MB of scratch

    cvt_f32_f16<<<2048, 256, 0, stream>>>(x,  x16,  (int)NX);
    cvt_f32_f16<<<1024, 256, 0, stream>>>(Wq, wq16, (int)NW);
    cvt_f32_f16<<<1024, 256, 0, stream>>>(Wk, wk16, (int)NW);
    cvt_f32_f16<<<1024, 256, 0, stream>>>(Wv, wv16, (int)NW);
    cvt_f32_f16<<<1024, 256, 0, stream>>>(Wo, wo16, (int)NW);

    proj_gemm<<<1024, 256, 0, stream>>>(x16, wq16, bq, q16);
    proj_gemm<<<1024, 256, 0, stream>>>(x16, wk16, bk, k16);
    proj_gemm<<<1024, 256, 0, stream>>>(x16, wv16, bv, v16);

    flash_attn<<<1024, 256, 0, stream>>>(q16, k16, v16, o16);

    out_gemm<<<1024, 256, 0, stream>>>(o16, wo16, bo, (float*)d_out);
}
